// DecoderBlock_26731876450822
// MI455X (gfx1250) — matrix operand order repaired and checked by reading, not compiled, not run on hardware
//
#include <hip/hip_runtime.h>
#include <cstddef>

typedef __attribute__((ext_vector_type(16))) __bf16 v16bf;
typedef __attribute__((ext_vector_type(8)))  float  v8f;
typedef __attribute__((ext_vector_type(4)))  unsigned int u32x4;
typedef __attribute__((ext_vector_type(4)))  int i32x4;
typedef __attribute__((ext_vector_type(8)))  int i32x8;

union AFrag { v16bf v; unsigned int u[8]; };
union BF8   { uint4 u; __bf16 b[8]; };

// Tensor Data Mover availability (arity differs across toolchains)
#if defined(__has_builtin)
#if __has_builtin(__builtin_amdgcn_tensor_load_to_lds) && __has_builtin(__builtin_amdgcn_s_wait_tensorcnt)
#define USE_TDM 1
#endif
#endif
#ifndef USE_TDM
#define USE_TDM 0
#endif

static __device__ __forceinline__ __bf16 f2bf(float f) {
    unsigned u = __builtin_bit_cast(unsigned, f);
    unsigned r = (u + 0x7FFFu + ((u >> 16) & 1u)) >> 16;
    unsigned short us = (unsigned short)r;
    return __builtin_bit_cast(__bf16, us);
}

#if USE_TDM
// Issue one TDM load of a 128(row) x 32(col) bf16 tile, row pitch K elements,
// into LDS at byte offset lds_off with row padding 8 bf16 (pitch 40).
static __device__ __forceinline__ void tdm_load_tile_A(const __bf16* gsrc, unsigned lds_off,
                                                       int K, int Mrows) {
    unsigned long long ga = (unsigned long long)(size_t)gsrc;
    u32x4 g0;
    g0[0] = 1u;                                                   // count=1, user mode
    g0[1] = lds_off;                                              // lds_addr (bytes)
    g0[2] = (unsigned)(ga & 0xFFFFFFFFull);                       // global_addr[31:0]
    g0[3] = (unsigned)((ga >> 32) & 0x1FFFFFFull) | (2u << 30);   // addr[56:32] | type=2
    i32x8 g1;
    g1[0] = (int)((1u << 16)      // data_size = 2 bytes
                | (1u << 20)      // pad_enable
                | (3u << 22)      // pad_interval: 16 DWORDs (one 32-elem row)
                | (3u << 25));    // pad_amount: 4 DWORDs (8 bf16)
    g1[1] = (int)(((unsigned)K & 0xFFFFu) << 16);                 // tensor_dim0[15:0]
    g1[2] = (int)(((unsigned)K >> 16) | (((unsigned)Mrows & 0xFFFFu) << 16));
    g1[3] = (int)(((unsigned)Mrows >> 16) | (32u << 16));         // tile_dim0 = 32
    g1[4] = 128;                                                  // tile_dim1=128, tile_dim2=0
    g1[5] = K;                                                    // tensor_dim0_stride[31:0]
    g1[6] = 0;
    g1[7] = 0;
    i32x4 gz = {};
#if __clang_major__ >= 23
    i32x8 gz8 = {};
    __builtin_amdgcn_tensor_load_to_lds(g0, g1, gz, gz, gz8, 0);
#else
    __builtin_amdgcn_tensor_load_to_lds(g0, g1, gz, gz, 0);
#endif
}
#endif

// ---------------------------------------------------------------------------
// fp32 -> bf16 elementwise convert
// ---------------------------------------------------------------------------
__global__ void cvt_bf16_kernel(const float* __restrict__ in, __bf16* __restrict__ out, int n) {
    int i = blockIdx.x * blockDim.x + threadIdx.x;
    int stride = gridDim.x * blockDim.x;
    for (; i < n; i += stride) out[i] = f2bf(in[i]);
}

// ---------------------------------------------------------------------------
// LayerNorm (D=1024) fp32 -> bf16, one block per row
// ---------------------------------------------------------------------------
__global__ __launch_bounds__(256)
void ln_bf16_kernel(const float* __restrict__ x, const float* __restrict__ w,
                    const float* __restrict__ b, __bf16* __restrict__ out) {
    const int row = blockIdx.x;
    const float* xr = x + (size_t)row * 1024;
    float s = 0.f, s2 = 0.f;
    for (int i = threadIdx.x; i < 1024; i += 256) { float v = xr[i]; s += v; s2 += v * v; }
    __shared__ float rs[256], rq[256];
    rs[threadIdx.x] = s; rq[threadIdx.x] = s2;
    __syncthreads();
    for (int off = 128; off > 0; off >>= 1) {
        if (threadIdx.x < off) { rs[threadIdx.x] += rs[threadIdx.x + off]; rq[threadIdx.x] += rq[threadIdx.x + off]; }
        __syncthreads();
    }
    float mu  = rs[0] * (1.0f / 1024.0f);
    float var = rq[0] * (1.0f / 1024.0f) - mu * mu;
    float inv = rsqrtf(var + 1e-5f);
    for (int i = threadIdx.x; i < 1024; i += 256)
        out[(size_t)row * 1024 + i] = f2bf((xr[i] - mu) * inv * w[i] + b[i]);
}

// ---------------------------------------------------------------------------
// bf16 GEMM:  C[Mrows,N] = act(A[Mrows,K] @ W[K,N] + bias)
// MODE 0: out bf16 ; MODE 1: out bf16 with exact GELU ; MODE 2: fp32 out +=
// Block tile 128x128, 8 waves (4x2), each wave 32x64 = 2x4 WMMA accumulators.
// Double-buffered LDS; A tile via Tensor Data Mover (wave 0), B tile via
// register prefetch + transposed store. One barrier per K-step.
// ---------------------------------------------------------------------------
template<int MODE>
__global__ __launch_bounds__(256)
void gemm_bf16_kernel(const __bf16* __restrict__ A, const __bf16* __restrict__ W,
                      const float* __restrict__ bias, __bf16* __restrict__ outB,
                      float* __restrict__ outF, int Mrows, int N, int K) {
    __shared__ __bf16 As[2][128][40];   // [m][k]
    __shared__ __bf16 Bs[2][128][40];   // [n][k] (transposed)
    const int t    = threadIdx.x;
    const int lane = t & 31;
    const int wave = t >> 5;
    const int half = lane >> 4;
    const int lr   = lane & 15;
    const int wm   = wave >> 1;      // 0..3
    const int wn   = wave & 1;       // 0..1
    const int m0   = blockIdx.y * 128;
    const int n0   = blockIdx.x * 128;

    v8f acc[2][4] = {};

    const int arow = t >> 1, acol = (t & 1) * 16;   // 128 rows x 32 cols of A
    const int brow = t >> 3, bcol = (t & 7) * 16;   // 32 rows x 128 cols of W

    // ---- prologue: stage k0 = 0 into buffer 0 ----
    {
        BF8 w0, w1;
        const uint4* wg = reinterpret_cast<const uint4*>(W + (size_t)brow * N + n0 + bcol);
        w0.u = wg[0]; w1.u = wg[1];
#pragma unroll
        for (int j = 0; j < 8; ++j) {
            Bs[0][bcol + j][brow]     = w0.b[j];
            Bs[0][bcol + 8 + j][brow] = w1.b[j];
        }
#if USE_TDM
        if (wave == 0) {
            tdm_load_tile_A(A + (size_t)m0 * K, (unsigned)(size_t)(void*)&As[0][0][0], K, Mrows);
            __builtin_amdgcn_s_wait_tensorcnt(0);
        }
#else
        const uint4* ag = reinterpret_cast<const uint4*>(A + (size_t)(m0 + arow) * K + acol);
        *reinterpret_cast<uint4*>(&As[0][arow][acol])     = ag[0];
        *reinterpret_cast<uint4*>(&As[0][arow][acol + 8]) = ag[1];
#endif
    }
    __syncthreads();

    int cur = 0;
    for (int k0 = 0; k0 < K; k0 += 32) {
        const int nxt = cur ^ 1;
        const bool has_next = (k0 + 32) < K;

        // prefetch next tiles while current tile computes
        BF8 w0, w1;
#if !USE_TDM
        uint4 a0 = {}, a1 = {};
#endif
        if (has_next) {
            const uint4* wg = reinterpret_cast<const uint4*>(W + (size_t)(k0 + 32 + brow) * N + n0 + bcol);
            w0.u = wg[0]; w1.u = wg[1];
#if USE_TDM
            if (wave == 0)
                tdm_load_tile_A(A + (size_t)m0 * K + (k0 + 32),
                                (unsigned)(size_t)(void*)&As[nxt][0][0], K, Mrows);
#else
            const uint4* ag = reinterpret_cast<const uint4*>(A + (size_t)(m0 + arow) * K + k0 + 32 + acol);
            a0 = ag[0]; a1 = ag[1];
#endif
        }

        // ---- compute on buffer `cur` ----
        AFrag af[2];
#pragma unroll
        for (int ms = 0; ms < 2; ++ms) {
            const int row = wm * 32 + ms * 16 + lr;
#pragma unroll
            for (int i = 0; i < 8; ++i) {
                const int kd = (i >> 2) * 16 + half * 8 + (i & 3) * 2;  // 16-bit A layout
                af[ms].u[i] = *reinterpret_cast<const unsigned int*>(&As[cur][row][kd]);
            }
        }
#pragma unroll
        for (int ns = 0; ns < 4; ++ns) {
            AFrag bfr;
            const int col = wn * 64 + ns * 16 + lr;
#pragma unroll
            for (int i = 0; i < 8; ++i) {
                const int kd = (i >> 2) * 16 + half * 8 + (i & 3) * 2;  // 16-bit B layout
                bfr.u[i] = *reinterpret_cast<const unsigned int*>(&Bs[cur][col][kd]);
            }
            acc[0][ns] = __builtin_amdgcn_wmma_f32_16x16x32_bf16(false, af[0].v, false, bfr.v,
                                                                 (short)0, acc[0][ns], false, false);
            acc[1][ns] = __builtin_amdgcn_wmma_f32_16x16x32_bf16(false, af[1].v, false, bfr.v,
                                                                 (short)0, acc[1][ns], false, false);
        }

        // ---- commit prefetched tiles into buffer `nxt` ----
        if (has_next) {
#pragma unroll
            for (int j = 0; j < 8; ++j) {
                Bs[nxt][bcol + j][brow]     = w0.b[j];
                Bs[nxt][bcol + 8 + j][brow] = w1.b[j];
            }
#if USE_TDM
            if (wave == 0) __builtin_amdgcn_s_wait_tensorcnt(0);
#else
            *reinterpret_cast<uint4*>(&As[nxt][arow][acol])     = a0;
            *reinterpret_cast<uint4*>(&As[nxt][arow][acol + 8]) = a1;
#endif
        }
        __syncthreads();
        cur = nxt;
    }

#pragma unroll
    for (int ms = 0; ms < 2; ++ms)
#pragma unroll
        for (int ns = 0; ns < 4; ++ns)
#pragma unroll
            for (int r = 0; r < 8; ++r) {
                const int grow = m0 + wm * 32 + ms * 16 + r + 8 * half;  // C/D layout
                const int gcol = n0 + wn * 64 + ns * 16 + lr;
                float v = acc[ms][ns][r] + bias[gcol];
                if (MODE == 1) v = 0.5f * v * (1.0f + erff(v * 0.70710678118f));
                if (MODE == 2) outF[(size_t)grow * N + gcol] += v;
                else           outB[(size_t)grow * N + gcol] = f2bf(v);
            }
}

// ---------------------------------------------------------------------------
// Flash attention, one wave per 16-query tile, 32-key blocks.
// Q:[B*Tq, qStride] (head col offset h*64), K/V:[B*Mk, *Stride].
// CAUSAL=1: causal mask; CAUSAL=0: mem_mask per (b,key).
// ---------------------------------------------------------------------------
template<int CAUSAL>
__global__ __launch_bounds__(128)
void attn_kernel(const __bf16* __restrict__ Qb, int qStride,
                 const __bf16* __restrict__ Kb, int kStride,
                 const __bf16* __restrict__ Vb, int vStride,
                 const int* __restrict__ mask,
                 __bf16* __restrict__ Yb, int yStride,
                 int Tq, int Mk, int Hh) {
    __shared__ __bf16 Plds[4][16][40];
    const int wid  = threadIdx.x >> 5;
    const int lane = threadIdx.x & 31;
    const int half = lane >> 4;
    const int lr   = lane & 15;
    const int bh = blockIdx.y;
    const int b  = bh / Hh;
    const int h  = bh % Hh;
    const int qrow0 = (blockIdx.x * 4 + wid) * 16;

    const __bf16* Qp = Qb + (size_t)(b * Tq) * qStride + h * 64;
    const __bf16* Kp = Kb + (size_t)(b * Mk) * kStride + h * 64;
    const __bf16* Vp = Vb + (size_t)(b * Mk) * vStride + h * 64;

    // Q fragments (16 rows x 64 K-dims = 2 A-frags), loaded once
    AFrag aq[2];
#pragma unroll
    for (int f = 0; f < 2; ++f)
#pragma unroll
        for (int i = 0; i < 8; ++i) {
            const int kd = f * 32 + (i >> 2) * 16 + half * 8 + (i & 3) * 2;
            aq[f].u[i] = *reinterpret_cast<const unsigned int*>(
                Qp + (size_t)(qrow0 + lr) * qStride + kd);
        }

    v8f o[4] = {};
    float mrun[8], lrun[8];
#pragma unroll
    for (int r = 0; r < 8; ++r) { mrun[r] = -1e30f; lrun[r] = 0.f; }

    const int nblk = CAUSAL ? ((qrow0 + 47) >> 5) : (Mk >> 5);

    for (int kb = 0; kb < nblk * 32; kb += 32) {
        // ---- S = Q @ K^T for 32 keys (two 16x16 tiles) ----
        v8f s0 = {}, s1 = {};
#pragma unroll
        for (int f = 0; f < 2; ++f) {
            AFrag bk;
#pragma unroll
            for (int i = 0; i < 8; ++i) {
                const int kd = f * 32 + (i >> 2) * 16 + half * 8 + (i & 3) * 2;
                bk.u[i] = *reinterpret_cast<const unsigned int*>(
                    Kp + (size_t)(kb + lr) * kStride + kd);
            }
            s0 = __builtin_amdgcn_wmma_f32_16x16x32_bf16(false, aq[f].v, false, bk.v,
                                                         (short)0, s0, false, false);
        }
#pragma unroll
        for (int f = 0; f < 2; ++f) {
            AFrag bk;
#pragma unroll
            for (int i = 0; i < 8; ++i) {
                const int kd = f * 32 + (i >> 2) * 16 + half * 8 + (i & 3) * 2;
                bk.u[i] = *reinterpret_cast<const unsigned int*>(
                    Kp + (size_t)(kb + 16 + lr) * kStride + kd);
            }
            s1 = __builtin_amdgcn_wmma_f32_16x16x32_bf16(false, aq[f].v, false, bk.v,
                                                         (short)0, s1, false, false);
        }

        bool mk0 = true, mk1 = true;
        if (!CAUSAL) {
            mk0 = mask[b * Mk + kb + lr] != 0;
            mk1 = mask[b * Mk + kb + 16 + lr] != 0;
        }

        // ---- online softmax (row = r + 8*half, col = lr in each half) ----
#pragma unroll
        for (int r = 0; r < 8; ++r) {
            float v0 = s0[r] * 0.125f;
            float v1 = s1[r] * 0.125f;
            bool ok0 = mk0, ok1 = mk1;
            if (CAUSAL) {
                const int qg = qrow0 + r + 8 * half;
                ok0 = (kb + lr) <= qg;
                ok1 = (kb + 16 + lr) <= qg;
            }
            if (!ok0) v0 = -1e30f;
            if (!ok1) v1 = -1e30f;
            float mx = fmaxf(v0, v1);
            mx = fmaxf(mx, __shfl_xor(mx, 1, 32));
            mx = fmaxf(mx, __shfl_xor(mx, 2, 32));
            mx = fmaxf(mx, __shfl_xor(mx, 4, 32));
            mx = fmaxf(mx, __shfl_xor(mx, 8, 32));
            const float mn  = fmaxf(mrun[r], mx);
            const float scl = __expf(mrun[r] - mn);
            mrun[r] = mn;
            const float e0 = ok0 ? __expf(v0 - mn) : 0.f;
            const float e1 = ok1 ? __expf(v1 - mn) : 0.f;
            float rsum = e0 + e1;
            rsum += __shfl_xor(rsum, 1, 32);
            rsum += __shfl_xor(rsum, 2, 32);
            rsum += __shfl_xor(rsum, 4, 32);
            rsum += __shfl_xor(rsum, 8, 32);
            lrun[r] = lrun[r] * scl + rsum;
            o[0][r] *= scl; o[1][r] *= scl; o[2][r] *= scl; o[3][r] *= scl;
            // transpose P into A-layout via per-wave LDS
            Plds[wid][r + 8 * half][lr]      = f2bf(e0);
            Plds[wid][r + 8 * half][lr + 16] = f2bf(e1);
        }
        asm volatile("s_wait_dscnt 0" ::: "memory");

        AFrag ap;
#pragma unroll
        for (int i = 0; i < 8; ++i) {
            const int kp = (i >> 2) * 16 + half * 8 + (i & 3) * 2;
            ap.u[i] = *reinterpret_cast<const unsigned int*>(&Plds[wid][lr][kp]);
        }

        // ---- O += P @ V ----
#pragma unroll
        for (int t = 0; t < 4; ++t) {
            AFrag bv;
#pragma unroll
            for (int i = 0; i < 8; ++i) {
                const int k0i = (i >> 2) * 16 + half * 8 + (i & 3) * 2;
                unsigned lo = *reinterpret_cast<const unsigned short*>(
                    Vp + (size_t)(kb + k0i) * vStride + t * 16 + lr);
                unsigned hi = *reinterpret_cast<const unsigned short*>(
                    Vp + (size_t)(kb + k0i + 1) * vStride + t * 16 + lr);
                bv.u[i] = lo | (hi << 16);
            }
            o[t] = __builtin_amdgcn_wmma_f32_16x16x32_bf16(false, ap.v, false, bv.v,
                                                           (short)0, o[t], false, false);
        }
    }

#pragma unroll
    for (int r = 0; r < 8; ++r) {
        const float inv = lrun[r] > 0.f ? 1.0f / lrun[r] : 0.f;
        const int grow = qrow0 + r + 8 * half;
#pragma unroll
        for (int t = 0; t < 4; ++t)
            Yb[(size_t)(b * Tq + grow) * yStride + h * 64 + t * 16 + lr] = f2bf(o[t][r] * inv);
    }
}

// ---------------------------------------------------------------------------
extern "C" void kernel_launch(void* const* d_in, const int* in_sizes, int n_in,
                              void* d_out, int out_size, void* d_ws, size_t ws_size,
                              hipStream_t stream) {
    (void)in_sizes; (void)n_in; (void)out_size; (void)ws_size;
    const int D = 1024, Hh = 16, DFF = 4096, B = 4, T = 2048, MM = 2048;
    const int BT = B * T, BM = B * MM;

    const float* x      = (const float*)d_in[0];
    const float* mem    = (const float*)d_in[1];
    const int*   mmask  = (const int*)d_in[2];
    const float* ln1_w  = (const float*)d_in[3];
    const float* ln1_b  = (const float*)d_in[4];
    const float* qkv_w  = (const float*)d_in[5];
    const float* qkv_b  = (const float*)d_in[6];
    const float* sap_w  = (const float*)d_in[7];
    const float* sap_b  = (const float*)d_in[8];
    const float* lnm_w  = (const float*)d_in[9];
    const float* lnm_b  = (const float*)d_in[10];
    const float* caq_w  = (const float*)d_in[11];
    const float* caq_b  = (const float*)d_in[12];
    const float* cakv_w = (const float*)d_in[13];
    const float* cakv_b = (const float*)d_in[14];
    const float* cap_w  = (const float*)d_in[15];
    const float* cap_b  = (const float*)d_in[16];
    const float* ln2_w  = (const float*)d_in[17];
    const float* ln2_b  = (const float*)d_in[18];
    const float* ff1_w  = (const float*)d_in[19];
    const float* ff1_b  = (const float*)d_in[20];
    const float* ff2_w  = (const float*)d_in[21];
    const float* ff2_b  = (const float*)d_in[22];
    float* out = (float*)d_out;

    char* base = (char*)d_ws;
    size_t off = 0;
    auto alloc = [&](size_t elems) -> __bf16* {
        __bf16* p = (__bf16*)(base + off);
        off += ((elems * sizeof(__bf16) + 255) / 256) * 256;
        return p;
    };
    __bf16* Wqkv  = alloc((size_t)D * 3 * D);
    __bf16* Wsap  = alloc((size_t)D * D);
    __bf16* Wcaq  = alloc((size_t)D * D);
    __bf16* Wcakv = alloc((size_t)D * 2 * D);
    __bf16* Wcap  = alloc((size_t)D * D);
    __bf16* Wff1  = alloc((size_t)D * DFF);
    __bf16* Wff2  = alloc((size_t)DFF * D);
    __bf16* Hbuf  = alloc((size_t)BT * D);
    __bf16* Big   = alloc((size_t)BT * DFF);
    __bf16* Qc    = alloc((size_t)BT * D);
    __bf16* Ybuf  = alloc((size_t)BT * D);
    __bf16* Membf = alloc((size_t)BM * D);

    // residual chain lives in d_out
    hipMemcpyAsync(out, x, (size_t)BT * D * sizeof(float), hipMemcpyDeviceToDevice, stream);

    auto cvt = [&](const float* src, __bf16* dst, size_t n) {
        cvt_bf16_kernel<<<dim3(2048), dim3(256), 0, stream>>>(src, dst, (int)n);
    };
    cvt(qkv_w,  Wqkv,  (size_t)D * 3 * D);
    cvt(sap_w,  Wsap,  (size_t)D * D);
    cvt(caq_w,  Wcaq,  (size_t)D * D);
    cvt(cakv_w, Wcakv, (size_t)D * 2 * D);
    cvt(cap_w,  Wcap,  (size_t)D * D);
    cvt(ff1_w,  Wff1,  (size_t)D * DFF);
    cvt(ff2_w,  Wff2,  (size_t)DFF * D);
    cvt(mem,    Membf, (size_t)BM * D);

    // ---- causal self-attention block ----
    ln_bf16_kernel<<<dim3(BT), dim3(256), 0, stream>>>(x, ln1_w, ln1_b, Hbuf);
    gemm_bf16_kernel<0><<<dim3(3 * D / 128, BT / 128), dim3(256), 0, stream>>>(
        Hbuf, Wqkv, qkv_b, Big, nullptr, BT, 3 * D, D);
    attn_kernel<1><<<dim3(T / 64, B * Hh), dim3(128), 0, stream>>>(
        Big, 3 * D, Big + D, 3 * D, Big + 2 * D, 3 * D, nullptr, Ybuf, D, T, T, Hh);
    gemm_bf16_kernel<2><<<dim3(D / 128, BT / 128), dim3(256), 0, stream>>>(
        Ybuf, Wsap, sap_b, nullptr, out, BT, D, D);

    // ---- cross-attention block ----
    ln_bf16_kernel<<<dim3(BT), dim3(256), 0, stream>>>(out, lnm_w, lnm_b, Hbuf);
    gemm_bf16_kernel<0><<<dim3(D / 128, BT / 128), dim3(256), 0, stream>>>(
        Hbuf, Wcaq, caq_b, Qc, nullptr, BT, D, D);
    gemm_bf16_kernel<0><<<dim3(2 * D / 128, BM / 128), dim3(256), 0, stream>>>(
        Membf, Wcakv, cakv_b, Big, nullptr, BM, 2 * D, D);
    attn_kernel<0><<<dim3(T / 64, B * Hh), dim3(128), 0, stream>>>(
        Qc, D, Big, 2 * D, Big + D, 2 * D, mmask, Ybuf, D, T, MM, Hh);
    gemm_bf16_kernel<2><<<dim3(D / 128, BT / 128), dim3(256), 0, stream>>>(
        Ybuf, Wcap, cap_b, nullptr, out, BT, D, D);

    // ---- FFN with exact GELU ----
    ln_bf16_kernel<<<dim3(BT), dim3(256), 0, stream>>>(out, ln2_w, ln2_b, Hbuf);
    gemm_bf16_kernel<1><<<dim3(DFF / 128, BT / 128), dim3(256), 0, stream>>>(
        Hbuf, Wff1, ff1_b, Big, nullptr, BT, DFF, D);
    gemm_bf16_kernel<2><<<dim3(D / 128, BT / 128), dim3(256), 0, stream>>>(
        Big, Wff2, ff2_b, nullptr, out, BT, D, DFF);
}
